// STMCModel_25769803776119
// MI455X (gfx1250) — compile-verified, matrix operand order checked by reading, not run on hardware
//
#include <hip/hip_runtime.h>
#include <hip/hip_bf16.h>
#include <cstdint>

// ---------------------------------------------------------------------------
// STMC seq2seq (biLSTM encoder x2 + LSTM decoder) for MI455X / gfx1250.
// - bf16 WMMA (v_wmma_f32_16x16x32_bf16) for every GEMM, f32 accumulators,
//   f32 LSTM cell state (the numerically sensitive recurrence).
// - Attention is algebraically degenerate (softmax over a length-1 axis == 1)
//   so context = sum_s enc_a[:,s,:], computed once.
// - The 16-row A strip of each tile row is contiguous (lda == K); it is staged
//   into LDS once per block via the Tensor Data Mover (tensor_load_to_lds +
//   s_wait_tensorcnt) so the 4-8 waves of a block stop re-fetching it, and
//   A fragments come from ds_load_b128.
// Workspace use: ~150 MB carved from d_ws.
// ---------------------------------------------------------------------------

typedef __bf16 bf16;
typedef __attribute__((ext_vector_type(16))) __bf16 bf16x16;
typedef __attribute__((ext_vector_type(8)))  float  f32x8;
typedef __attribute__((ext_vector_type(4)))  unsigned int u32x4;
typedef __attribute__((ext_vector_type(4)))  int    i32x4;
typedef __attribute__((ext_vector_type(8)))  int    i32x8;

#ifndef __has_builtin
#define __has_builtin(x) 0
#endif
#if __has_builtin(__builtin_amdgcn_tensor_load_to_lds) && \
    __has_builtin(__builtin_amdgcn_s_wait_tensorcnt)
#define USE_TDM 1
#else
#define USE_TDM 0
#endif

// Load a 16x32 bf16 A/B fragment from a row-major [rows, ld] matrix (global or
// LDS). ISA layout (CDNA5 ISA): lanes 0-15 hold row (lane), K in
// {0..7, 16..23}; lanes 16-31 hold row (lane-16), K in {8..15, 24..31}.
__device__ __forceinline__ bf16x16 load_frag(const bf16* p, int ld) {
  const int lane = threadIdx.x & 31;
  const int r    = lane & 15;
  const int hk   = (lane >> 4) << 3;   // 0 or 8
  const bf16* q  = p + (long)r * ld + hk;
  union { bf16x16 v; uint4 u[2]; } t;
  t.u[0] = *(const uint4*)(q);         // K chunk +0..7
  t.u[1] = *(const uint4*)(q + 16);    // K chunk +16..23
  return t.v;
}

#if USE_TDM
// Issue a 1-D TDM transfer of `nbytes` (<= 65535, multiple of 16) from global
// `gsrc` to LDS `lds_dst`. D# packing per CDNA5 ISA:
//   group0: [1:0]=count=1, [63:32]=lds_addr, [120:64]=global_addr, [127:126]=2
//   group1: data_size=0 (bytes); tensor_dim0 = tile_dim0 = dim0_stride = nbytes
//           tensor_dim1 = tile_dim1 = 1 (single row)
__device__ __forceinline__ void tdm_load_1d(void* lds_dst, const void* gsrc,
                                            unsigned int nbytes) {
  const unsigned long long ga = (unsigned long long)(uintptr_t)gsrc;
  const unsigned int ldsa = (unsigned int)(uintptr_t)lds_dst; // LDS byte offset
  u32x4 g0;
  g0[0] = 1u;                                   // count=1, user mode
  g0[1] = ldsa;                                 // lds_addr
  g0[2] = (unsigned int)ga;                     // global_addr[31:0]
  g0[3] = (unsigned int)((ga >> 32) & 0x01FFFFFFu) | (2u << 30); // addr|type=2
  i32x8 g1;
  g1[0] = 0;                                    // wg_mask=0, data_size=1B
  g1[1] = (int)((nbytes & 0xFFFFu) << 16);      // tensor_dim0[15:0] @ bit48
  g1[2] = (int)(((nbytes >> 16) & 0xFFFFu) | (1u << 16)); // dim0 hi | dim1=1
  g1[3] = (int)((nbytes & 0xFFFFu) << 16);      // tile_dim0 @ bit112
  g1[4] = 1;                                    // tile_dim1 = 1
  g1[5] = (int)nbytes;                          // tensor_dim0_stride lo
  g1[6] = 0;
  g1[7] = 0;
  i32x4 z4 = {0, 0, 0, 0};
#if defined(__clang_major__) && __clang_major__ >= 23
  i32x8 z8 = {0, 0, 0, 0, 0, 0, 0, 0};
  __builtin_amdgcn_tensor_load_to_lds(g0, g1, z4, z4, z8, 0);
#else
  __builtin_amdgcn_tensor_load_to_lds(g0, g1, z4, z4, 0);
#endif
}
#endif  // USE_TDM

// Stage `nbytes` contiguous bytes global -> LDS (block-cooperative fallback).
__device__ __forceinline__ void stage_copy(void* dst, const void* src,
                                           unsigned int nbytes) {
  const uint4* s = (const uint4*)src;
  uint4* d = (uint4*)dst;
  const int n = (int)(nbytes >> 4);
  for (int i = threadIdx.x; i < n; i += blockDim.x) d[i] = s[i];
}

__device__ __forceinline__ float sigm(float x) { return 1.0f / (1.0f + __expf(-x)); }

// ----------------------------- utility kernels -----------------------------

__global__ void k_cvt(const float* __restrict__ s, bf16* __restrict__ d, int n) {
  int i = blockIdx.x * blockDim.x + threadIdx.x;
  if (i < n) d[i] = (bf16)s[i];
}
__global__ void k_zero_b16(bf16* p, int n) {
  int i = blockIdx.x * blockDim.x + threadIdx.x;
  if (i < n) p[i] = (bf16)0.0f;
}
__global__ void k_zero_f32(float* p, int n) {
  int i = blockIdx.x * blockDim.x + threadIdx.x;
  if (i < n) p[i] = 0.0f;
}
// strided 2D copies: n = rows*cols, row r = i/cols, col c = i%cols
__global__ void k_cp_b16(const bf16* __restrict__ s, int sld,
                         bf16* __restrict__ d, int dld, int cols, int n) {
  int i = blockIdx.x * blockDim.x + threadIdx.x;
  if (i >= n) return;
  int r = i / cols, c = i % cols;
  d[(long)r * dld + c] = s[(long)r * sld + c];
}
__global__ void k_cp_f32(const float* __restrict__ s, int sld,
                         float* __restrict__ d, int dld, int cols, int n) {
  int i = blockIdx.x * blockDim.x + threadIdx.x;
  if (i >= n) return;
  int r = i / cols, c = i % cols;
  d[(long)r * dld + c] = s[(long)r * sld + c];
}
// context[b,j] = sum_t y[b,t,j]  (y is [B, S, C] bf16, f32 accumulate)
__global__ void k_context(const bf16* __restrict__ y, bf16* __restrict__ ctx,
                          int S_, int C, int n) {
  int i = blockIdx.x * blockDim.x + threadIdx.x;
  if (i >= n) return;
  int b = i / C, j = i % C;
  const bf16* p = y + (long)b * S_ * C + j;
  float s = 0.0f;
  for (int t = 0; t < S_; ++t) s += (float)p[(long)t * C];
  ctx[i] = (bf16)s;
}

// ------------------------------- GEMM kernel -------------------------------
// C[M,N](f32) = A[M,K](bf16) * W[N,K](bf16)^T + bias[N]; optional bf16 copy.
// One wave per 16x16 tile; block = 8 waves (128 output cols). The shared
// 16-row A strip (contiguous, 32*K bytes) is staged into LDS via TDM.
__global__ void k_gemm(const bf16* __restrict__ A, const bf16* __restrict__ W,
                       const float* __restrict__ bias,
                       float* __restrict__ C, long ldc,
                       bf16* __restrict__ Cb, int ldcb, int K) {
  extern __shared__ char smem[];
  bf16* sA = (bf16*)smem;
  const int wave = threadIdx.x >> 5;
  const int n0   = (blockIdx.x * (blockDim.x >> 5) + wave) * 16;
  const int m0   = blockIdx.y * 16;
  const int lane = threadIdx.x & 31;
  const int col  = lane & 15;
  const int rb   = (lane >> 4) * 8;
  const unsigned int strip_bytes = (unsigned int)(32 * K);

#if USE_TDM
  if (threadIdx.x == 0) {
    tdm_load_1d(sA, A + (long)m0 * K, strip_bytes);
    __builtin_amdgcn_s_wait_tensorcnt(0);
  }
#else
  stage_copy(sA, A + (long)m0 * K, strip_bytes);
#endif
  __syncthreads();

  f32x8 acc;
  const float bv = bias ? bias[n0 + col] : 0.0f;
#pragma unroll
  for (int i = 0; i < 8; ++i) acc[i] = bv;

  for (int k = 0; k < K; k += 32) {
    bf16x16 a = load_frag(sA + k, K);
    bf16x16 w = load_frag(W + (long)n0 * K + k, K);
    acc = __builtin_amdgcn_wmma_f32_16x16x32_bf16(false, a, false, w,
                                                  (short)0, acc, false, false);
  }
#pragma unroll
  for (int i = 0; i < 8; ++i) {
    const long row = m0 + rb + i;
    C[row * ldc + n0 + col] = acc[i];
    if (Cb) Cb[row * (long)ldcb + n0 + col] = (bf16)acc[i];
  }
}

// ------------------------------ LSTM cell kernel ---------------------------
// gates = (xg[t] | bias) + A1*W1^T (+ A2*W2^T)   [torch gate order i,f,g,o]
// c' = sigm(f)*c + sigm(i)*tanh(g);  h' = sigm(o)*tanh(c')
// Each wave owns one 16(batch) x 16(hidden) tile and accumulates all 4 gate
// tiles (columns n0 + g*Hc) reusing the A fragment, fusing the nonlinearity
// in-register. A strips (16 x K1 [+ 16 x K2], contiguous) staged in LDS.
__global__ void k_lstm_cell(
    const float* __restrict__ xg, int ldxg,                 // [B,4Hc] or null
    const bf16* __restrict__ A1, const bf16* __restrict__ W1, int K1,
    const bf16* __restrict__ A2, const bf16* __restrict__ W2, int K2,
    const float* __restrict__ bias,                         // [4Hc] or null
    const float* __restrict__ c_in, float* __restrict__ c_out,  // [B,Hc] f32
    bf16* __restrict__ h_out,                               // [B,Hc] bf16
    bf16* __restrict__ y_out, int ldy,                      // optional seq out
    int Hc) {
  extern __shared__ char smem[];
  bf16* sA1 = (bf16*)smem;
  bf16* sA2 = (bf16*)(smem + (size_t)32 * K1);
  const int wave = threadIdx.x >> 5;
  const int n0   = (blockIdx.x * (blockDim.x >> 5) + wave) * 16;
  const int m0   = blockIdx.y * 16;
  const int lane = threadIdx.x & 31;
  const int col  = lane & 15;
  const int rb   = (lane >> 4) * 8;

#if USE_TDM
  if (threadIdx.x == 0) {
    tdm_load_1d(sA1, A1 + (long)m0 * K1, (unsigned int)(32 * K1));
    if (K2 > 0) tdm_load_1d(sA2, A2 + (long)m0 * K2, (unsigned int)(32 * K2));
    __builtin_amdgcn_s_wait_tensorcnt(0);
  }
#else
  stage_copy(sA1, A1 + (long)m0 * K1, (unsigned int)(32 * K1));
  if (K2 > 0) stage_copy(sA2, A2 + (long)m0 * K2, (unsigned int)(32 * K2));
#endif
  __syncthreads();

  f32x8 acc[4];
#pragma unroll
  for (int g = 0; g < 4; ++g) {
    const float bv = bias ? bias[g * Hc + n0 + col] : 0.0f;
#pragma unroll
    for (int i = 0; i < 8; ++i) {
      float v = bv;
      if (xg) v += xg[(long)(m0 + rb + i) * ldxg + g * Hc + n0 + col];
      acc[g][i] = v;
    }
  }
  for (int k = 0; k < K1; k += 32) {
    bf16x16 a = load_frag(sA1 + k, K1);
#pragma unroll
    for (int g = 0; g < 4; ++g) {
      bf16x16 w = load_frag(W1 + (long)(g * Hc + n0) * K1 + k, K1);
      acc[g] = __builtin_amdgcn_wmma_f32_16x16x32_bf16(false, a, false, w,
                                                       (short)0, acc[g], false, false);
    }
  }
  if (K2 > 0) {
    for (int k = 0; k < K2; k += 32) {
      bf16x16 a = load_frag(sA2 + k, K2);
#pragma unroll
      for (int g = 0; g < 4; ++g) {
        bf16x16 w = load_frag(W2 + (long)(g * Hc + n0) * K2 + k, K2);
        acc[g] = __builtin_amdgcn_wmma_f32_16x16x32_bf16(false, a, false, w,
                                                         (short)0, acc[g], false, false);
      }
    }
  }
#pragma unroll
  for (int i = 0; i < 8; ++i) {
    const long row = m0 + rb + i;
    const long idx = row * Hc + n0 + col;
    const float iv = sigm(acc[0][i]);
    const float fv = sigm(acc[1][i]);
    const float gv = tanhf(acc[2][i]);
    const float ov = sigm(acc[3][i]);
    const float c  = fv * c_in[idx] + iv * gv;
    const float h  = ov * tanhf(c);
    c_out[idx] = c;
    h_out[idx] = (bf16)h;
    if (y_out) y_out[row * (long)ldy + n0 + col] = (bf16)h;
  }
}

// --------------------------------- driver ----------------------------------

static inline int cdiv(long a, int b) { return (int)((a + b - 1) / b); }

extern "C" void kernel_launch(void* const* d_in, const int* in_sizes, int n_in,
                              void* d_out, int out_size, void* d_ws, size_t ws_size,
                              hipStream_t stream) {
  (void)in_sizes; (void)n_in; (void)ws_size;

  const float* sign_a   = (const float*)d_in[0];
  const float* sign_b   = (const float*)d_in[1];
  const float* enc_Wih0 = (const float*)d_in[2];
  const float* enc_Whh0 = (const float*)d_in[3];
  const float* enc_b0   = (const float*)d_in[4];
  const float* enc_Wih1 = (const float*)d_in[5];
  const float* enc_Whh1 = (const float*)d_in[6];
  const float* enc_b1   = (const float*)d_in[7];
  const float* dec_Wih0 = (const float*)d_in[8];
  const float* dec_Whh0 = (const float*)d_in[9];
  const float* dec_b0   = (const float*)d_in[10];
  const float* dec_Wih1 = (const float*)d_in[11];
  const float* dec_Whh1 = (const float*)d_in[12];
  const float* dec_b1   = (const float*)d_in[13];
  // d_in[14]/d_in[15] (attn_W/attn_b) unused: softmax over length-1 axis == 1.
  const float* out_Wp   = (const float*)d_in[16];
  const float* out_bp   = (const float*)d_in[17];

  const int B = 64, S = 64, E = 256, H = 512;
  const int G = 4 * H;        // 2048, encoder gate width
  const int Hd = 2 * H;       // 1024, decoder hidden
  const int J4 = 512;
  const int T = out_size / (B * J4);   // transition_length (=30)

  // ---- carve workspace (256B aligned) ----
  char* base = (char*)d_ws; size_t off = 0;
  auto alloc = [&](size_t bytes) -> void* {
    void* r = base + off; off += (bytes + 255) & ~(size_t)255; return r;
  };
  bf16* Wih0b  = (bf16*)alloc((size_t)2 * G * E * 2);
  bf16* Whh0b  = (bf16*)alloc((size_t)2 * G * H * 2);
  bf16* Wih1b  = (bf16*)alloc((size_t)2 * G * (2 * H) * 2);
  bf16* Whh1b  = (bf16*)alloc((size_t)2 * G * H * 2);
  bf16* dWih0b = (bf16*)alloc((size_t)4 * Hd * (3 * H) * 2);
  bf16* dWhh0b = (bf16*)alloc((size_t)4 * Hd * Hd * 2);
  bf16* dWih1b = (bf16*)alloc((size_t)4 * Hd * Hd * 2);
  bf16* dWhh1b = (bf16*)alloc((size_t)4 * Hd * Hd * 2);
  bf16* outWb  = (bf16*)alloc((size_t)J4 * Hd * 2);
  bf16* xab    = (bf16*)alloc((size_t)B * S * E * 2);
  bf16* xbb    = (bf16*)alloc((size_t)B * S * E * 2);
  float* xg    = (float*)alloc((size_t)2 * B * S * G * 4);   // [dir][B*S][4H]
  bf16* y0b    = (bf16*)alloc((size_t)B * S * (2 * H) * 2);
  bf16* y1b    = (bf16*)alloc((size_t)B * S * (2 * H) * 2);
  bf16* hpp    = (bf16*)alloc((size_t)2 * 2 * B * H * 2);    // [dir][pp][B*H]
  float* cpp   = (float*)alloc((size_t)2 * 2 * B * H * 4);
  bf16* dh0    = (bf16*)alloc((size_t)2 * B * Hd * 2);       // decoder pp
  bf16* dh1    = (bf16*)alloc((size_t)2 * B * Hd * 2);
  float* dc0   = (float*)alloc((size_t)2 * B * Hd * 4);
  float* dc1   = (float*)alloc((size_t)2 * B * Hd * 4);
  bf16* ctxb   = (bf16*)alloc((size_t)B * Hd * 2);
  bf16* kfb    = (bf16*)alloc((size_t)B * J4 * 2);
  bf16* xbuf   = (bf16*)alloc((size_t)B * (3 * H) * 2);

  // ---- weight/input conversion to bf16 ----
  auto cvt = [&](const float* s, bf16* d, long n) {
    k_cvt<<<cdiv(n, 256), 256, 0, stream>>>(s, d, (int)n);
  };
  cvt(enc_Wih0, Wih0b, 2L * G * E);
  cvt(enc_Whh0, Whh0b, 2L * G * H);
  cvt(enc_Wih1, Wih1b, 2L * G * 2 * H);
  cvt(enc_Whh1, Whh1b, 2L * G * H);
  cvt(dec_Wih0, dWih0b, 4L * Hd * 3 * H);
  cvt(dec_Whh0, dWhh0b, 4L * Hd * Hd);
  cvt(dec_Wih1, dWih1b, 4L * Hd * Hd);
  cvt(dec_Whh1, dWhh1b, 4L * Hd * Hd);
  cvt(out_Wp,   outWb,  (long)J4 * Hd);
  cvt(sign_a,   xab,    (long)B * S * E);
  cvt(sign_b,   xbb,    (long)B * S * E);

  auto run_cell = [&](const float* xg_t, int ldxg,
                      const bf16* A1, const bf16* W1, int K1,
                      const bf16* A2, const bf16* W2, int K2,
                      const float* bias, const float* cin, float* cout,
                      bf16* hout, bf16* yout, int ldy, int Hc) {
    dim3 grid(Hc / 64, B / 16);
    const size_t sh = (size_t)32 * (K1 + K2);   // LDS for A strips
    k_lstm_cell<<<grid, 128, sh, stream>>>(xg_t, ldxg, A1, W1, K1, A2, W2, K2,
                                           bias, cin, cout, hout, yout, ldy, Hc);
  };

  // ---- encoder: 2 biLSTM layers; xg precomputed, 64 sequential cell steps ----
  auto run_encode = [&](const bf16* xin) {
    for (int layer = 0; layer < 2; ++layer) {
      const bf16* Ain  = layer ? y0b : xin;
      const int   K    = layer ? 2 * H : E;
      const bf16* Wih  = layer ? Wih1b : Wih0b;
      const bf16* Whh  = layer ? Whh1b : Whh0b;
      const float* bv  = layer ? enc_b1 : enc_b0;
      bf16* yo         = layer ? y1b : y0b;
      for (int dir = 0; dir < 2; ++dir) {
        k_gemm<<<dim3(G / 128, (B * S) / 16), 256, (size_t)32 * K, stream>>>(
            Ain, Wih + (size_t)dir * G * K, bv + dir * G,
            xg + (size_t)dir * B * S * G, (long)G, nullptr, 0, K);
        k_zero_b16<<<cdiv(B * H, 256), 256, 0, stream>>>(
            hpp + ((size_t)dir * 2 + 0) * B * H, B * H);
        k_zero_f32<<<cdiv(B * H, 256), 256, 0, stream>>>(
            cpp + ((size_t)dir * 2 + 0) * B * H, B * H);
      }
      for (int t = 0; t < S; ++t) {
        for (int dir = 0; dir < 2; ++dir) {
          const int te = dir ? (S - 1 - t) : t;
          run_cell(xg + (size_t)dir * B * S * G + (size_t)te * G, S * G,
                   hpp + ((size_t)dir * 2 + (t & 1)) * B * H,
                   Whh + (size_t)dir * G * H, H,
                   nullptr, nullptr, 0,
                   nullptr,  // bias already folded into xg
                   cpp + ((size_t)dir * 2 + (t & 1)) * B * H,
                   cpp + ((size_t)dir * 2 + ((t + 1) & 1)) * B * H,
                   hpp + ((size_t)dir * 2 + ((t + 1) & 1)) * B * H,
                   yo + (size_t)te * (2 * H) + (size_t)dir * H, S * 2 * H, H);
        }
      }
    }
  };

  // After S steps the final state sits in ping-pong slot S&1 == 0.
  const bf16*  h_fin = hpp + ((size_t)1 * 2 + 0) * B * H;   // backward dir
  const float* c_fin = cpp + ((size_t)1 * 2 + 0) * B * H;

  // ---------- encode A ----------
  run_encode(xab);
  k_context<<<cdiv((long)B * Hd, 256), 256, 0, stream>>>(y1b, ctxb, S, 2 * H, B * Hd);
  // x0[:, 0:2H] = enc_a[:, -1, :]
  k_cp_b16<<<cdiv((long)B * 2 * H, 256), 256, 0, stream>>>(
      y1b + (size_t)(S - 1) * (2 * H), S * 2 * H, xbuf, 3 * H, 2 * H, B * 2 * H);
  // decoder init (first half = encode-A backward finals), replicated to layer 1
  k_cp_b16<<<cdiv(B * H, 256), 256, 0, stream>>>(h_fin, H, dh0, Hd, H, B * H);
  k_cp_b16<<<cdiv(B * H, 256), 256, 0, stream>>>(h_fin, H, dh1, Hd, H, B * H);
  k_cp_f32<<<cdiv(B * H, 256), 256, 0, stream>>>(c_fin, H, dc0, Hd, H, B * H);
  k_cp_f32<<<cdiv(B * H, 256), 256, 0, stream>>>(c_fin, H, dc1, Hd, H, B * H);

  // ---------- encode B ----------
  run_encode(xbb);
  // x0[:, 2H:3H] = enc_b[:, 0, :H] (forward half of first step)
  k_cp_b16<<<cdiv(B * H, 256), 256, 0, stream>>>(
      y1b, S * 2 * H, xbuf + 2 * H, 3 * H, H, B * H);
  // decoder init second halves (encode-B backward finals)
  k_cp_b16<<<cdiv(B * H, 256), 256, 0, stream>>>(h_fin, H, dh0 + H, Hd, H, B * H);
  k_cp_b16<<<cdiv(B * H, 256), 256, 0, stream>>>(h_fin, H, dh1 + H, Hd, H, B * H);
  k_cp_f32<<<cdiv(B * H, 256), 256, 0, stream>>>(c_fin, H, dc0 + H, Hd, H, B * H);
  k_cp_f32<<<cdiv(B * H, 256), 256, 0, stream>>>(c_fin, H, dc1 + H, Hd, H, B * H);

  // ---------- decoder ----------
  for (int t = 0; t < T; ++t) {
    const int p0 = t & 1, p1 = p0 ^ 1;
    // layer 0: gates = x*Wih0^T + h*Whh0^T + b0  (K chains fused in one kernel)
    run_cell(nullptr, 0,
             dh0 + (size_t)p0 * B * Hd, dWhh0b, Hd,
             xbuf, dWih0b, 3 * H,
             dec_b0,
             dc0 + (size_t)p0 * B * Hd, dc0 + (size_t)p1 * B * Hd,
             dh0 + (size_t)p1 * B * Hd, nullptr, 0, Hd);
    // layer 1
    run_cell(nullptr, 0,
             dh1 + (size_t)p0 * B * Hd, dWhh1b, Hd,
             dh0 + (size_t)p1 * B * Hd, dWih1b, Hd,
             dec_b1,
             dc1 + (size_t)p0 * B * Hd, dc1 + (size_t)p1 * B * Hd,
             dh1 + (size_t)p1 * B * Hd, nullptr, 0, Hd);
    // keyframe = h1 * out_W^T + out_b -> d_out[:, t, :] (f32) and kfb (bf16)
    k_gemm<<<dim3(J4 / 128, B / 16), 256, (size_t)32 * Hd, stream>>>(
        dh1 + (size_t)p1 * B * Hd, outWb, out_bp,
        (float*)d_out + (size_t)t * J4, (long)T * J4, kfb, J4, Hd);
    // x_{t+1} = [context (2H), keyframe (J4)]
    k_cp_b16<<<cdiv((long)B * Hd, 256), 256, 0, stream>>>(ctxb, Hd, xbuf, 3 * H, Hd, B * Hd);
    k_cp_b16<<<cdiv((long)B * J4, 256), 256, 0, stream>>>(kfb, J4, xbuf + 2 * H, 3 * H, J4, B * J4);
  }
}
